// InterestDict_71511205478455
// MI455X (gfx1250) — compile-verified, matrix operand order checked by reading, not run on hardware
//
#include <hip/hip_runtime.h>
#include <hip/hip_bf16.h>

// ---------- types for WMMA ----------
typedef __attribute__((ext_vector_type(16))) __bf16   v16bf;
typedef __attribute__((ext_vector_type(8)))  float    v8f;
typedef __attribute__((ext_vector_type(4)))  unsigned int u32x4;

union FragBF {
  u32x4 u[2];   // 2 x 16 bytes = 16 bf16
  v16bf v;
};

#define C_DIM 512
#define D_DIM 256
#define KTOP  4
#define MROWS 32   // X rows per block (2 M-tiles of 16)

__device__ __forceinline__ unsigned short f32_to_bf16_rne(float f) {
  unsigned int u = __float_as_uint(f);
  u += 0x7FFFu + ((u >> 16) & 1u);   // round-to-nearest-even
  return (unsigned short)(u >> 16);
}

// ---------- kernel 1: dict -> bf16, d2[c] = ||dict_c||^2 ----------
__global__ __launch_bounds__(256)
void prep_dict_kernel(const float* __restrict__ dict,
                      unsigned short* __restrict__ dictb,
                      float* __restrict__ d2) {
  __shared__ float red[256];
  const int c = blockIdx.x;
  const int t = threadIdx.x;
  float v = dict[c * D_DIM + t];
  dictb[c * D_DIM + t] = f32_to_bf16_rne(v);
  red[t] = v * v;
  __syncthreads();
  for (int s = 128; s > 0; s >>= 1) {
    if (t < s) red[t] += red[t + s];
    __syncthreads();
  }
  if (t == 0) d2[c] = red[0];
}

// ---------- kernel 2: WMMA distances + top-4 + group_emb + scatter ----------
__global__ __launch_bounds__(256)
void assign_kernel(const float* __restrict__ X,
                   const float* __restrict__ dictf,
                   const unsigned short* __restrict__ dictb,
                   const float* __restrict__ d2,
                   float* __restrict__ out_ge,
                   float* __restrict__ out_tk,
                   float* __restrict__ cs_accum,
                   float* __restrict__ in_sum) {
  __shared__ unsigned short Ash[MROWS * D_DIM];  // 16 KB: X tile as bf16
  __shared__ float Dist[MROWS * C_DIM];          // 64 KB: score tile
  __shared__ int   Tk[MROWS][KTOP];

  const int tid   = threadIdx.x;
  const int nBase = blockIdx.x * MROWS;

  // Phase 1: stage 32x256 X tile into LDS as bf16
  for (int e = tid; e < MROWS * D_DIM; e += 256) {
    const int r = e >> 8, col = e & 255;
    Ash[e] = f32_to_bf16_rne(X[(size_t)(nBase + r) * D_DIM + col]);
  }
  __syncthreads();

  // Phase 2: score = ||d_c||^2 - 2 * <x_n, d_c> via v_wmma_f32_16x16x32_bf16
  // Each wave: 2 M-tiles x 4 N-tiles; each B fragment feeds 2 wmmas.
  const int lane  = tid & 31;
  const int wave  = tid >> 5;
  const int l15   = lane & 15;
  const int hi    = lane >> 4;          // K half select (ISA 16-bit layout)
  const int cbase = wave * 64;          // 8 waves x 64 clusters = 512

  v8f acc[2][4] = {};
  const u32x4* arow0 = (const u32x4*)&Ash[(l15)      * D_DIM];
  const u32x4* arow1 = (const u32x4*)&Ash[(16 + l15) * D_DIM];

  #pragma unroll
  for (int kk = 0; kk < 8; ++kk) {      // K = 256 in steps of 32
    const int kq = kk * 4;              // 16B-chunk index within a row
    FragBF a0, a1;
    a0.u[0] = arow0[kq + hi];  a0.u[1] = arow0[kq + hi + 2];
    a1.u[0] = arow1[kq + hi];  a1.u[1] = arow1[kq + hi + 2];
    #pragma unroll
    for (int t = 0; t < 4; ++t) {
      const int c = cbase + t * 16 + l15;
      const u32x4* brow = (const u32x4*)&dictb[(size_t)c * D_DIM];
      FragBF b;
      b.u[0] = brow[kq + hi];
      b.u[1] = brow[kq + hi + 2];
      acc[0][t] = __builtin_amdgcn_wmma_f32_16x16x32_bf16(
          false, a0.v, false, b.v, (short)0, acc[0][t], false, false);
      acc[1][t] = __builtin_amdgcn_wmma_f32_16x16x32_bf16(
          false, a1.v, false, b.v, (short)0, acc[1][t], false, false);
    }
  }

  // C/D layout: VGPR r, lanes 0-15 -> M=r, N=lane; lanes 16-31 -> M=r+8
  #pragma unroll
  for (int t = 0; t < 4; ++t) {
    const int c = cbase + t * 16 + l15;
    const float dd = d2[c];
    #pragma unroll
    for (int mt = 0; mt < 2; ++mt) {
      #pragma unroll
      for (int r = 0; r < 8; ++r) {
        const int m = mt * 16 + r + hi * 8;
        Dist[m * C_DIM + c] = dd - 2.0f * acc[mt][t][r];
      }
    }
  }
  __syncthreads();

  // Phase 3: top-4 smallest per row (32 rows, one thread each)
  if (tid < MROWS) {
    const float* drow = &Dist[tid * C_DIM];
    float v0 = 3.4e38f, v1 = 3.4e38f, v2 = 3.4e38f, v3 = 3.4e38f;
    int   i0 = 0, i1 = 0, i2 = 0, i3 = 0;
    for (int c = 0; c < C_DIM; ++c) {
      const float val = drow[c];
      if (val < v3) {
        v3 = val; i3 = c;
        if (v3 < v2) { float tv=v3; v3=v2; v2=tv; int ti=i3; i3=i2; i2=ti;
          if (v2 < v1) { tv=v2; v2=v1; v1=tv; ti=i2; i2=i1; i1=ti;
            if (v1 < v0) { tv=v1; v1=v0; v0=tv; ti=i1; i1=i0; i0=ti; } } }
      }
    }
    Tk[tid][0] = i0; Tk[tid][1] = i1; Tk[tid][2] = i2; Tk[tid][3] = i3;
    float* tkout = &out_tk[(size_t)(nBase + tid) * KTOP];
    tkout[0] = (float)i0; tkout[1] = (float)i1;
    tkout[2] = (float)i2; tkout[3] = (float)i3;
  }
  __syncthreads();

  // Phase 4: group_emb = mean of top-4 dict rows (f32 dict; STE => identity)
  for (int e = tid; e < MROWS * D_DIM; e += 256) {
    const int r = e >> 8, col = e & 255;
    float s = 0.0f;
    #pragma unroll
    for (int p = 0; p < KTOP; ++p)
      s += dictf[(size_t)Tk[r][p] * D_DIM + col];
    out_ge[(size_t)(nBase + r) * D_DIM + col] = 0.25f * s;
  }

  // Phase 5: scatter-accumulate cluster_size and input_sum (f32 atomics)
  if (tid < MROWS) {
    #pragma unroll
    for (int p = 0; p < KTOP; ++p)
      atomicAdd(&cs_accum[Tk[tid][p]], 1.0f);
  }
  for (int e = tid; e < MROWS * KTOP * D_DIM; e += 256) {
    const int col = e & 255;
    const int p   = (e >> 8) & 3;
    const int r   = e >> 10;
    const float xv = X[(size_t)(nBase + r) * D_DIM + col];
    atomicAdd(&in_sum[(size_t)Tk[r][p] * D_DIM + col], xv);
  }
}

// ---------- kernel 3: sequential decay scan + EMA + laplace + normalize ----------
__global__ __launch_bounds__(256)
void finalize_kernel(const float* __restrict__ css_in,
                     const float* __restrict__ cv_in,
                     const float* __restrict__ cun_in,
                     const float* __restrict__ cs_accum,
                     const float* __restrict__ in_sum,
                     float* __restrict__ out_dict,
                     float* __restrict__ out_css,
                     float* __restrict__ out_cv,
                     float* __restrict__ out_cun,
                     float* __restrict__ decays) {
  __shared__ float shS;
  const int tid = threadIdx.x;

  if (tid == 0) {
    // faithful port of the lax.scan: decay carried across clusters
    float decay = 0.99f;
    float S = 0.0f;
    for (int c = 0; c < C_DIM; ++c) {
      const float size = cs_accum[c];
      const bool  m    = (size != 0.0f);
      const float un   = cun_in[c] + (m ? 1.0f : 0.0f);
      out_cun[c] = un;
      if (m && un <= 10.0f) decay = decay * un * 0.1f;
      decays[c] = decay;
      const float css  = css_in[c];
      const float cssn = m ? (css * decay + size * (1.0f - decay)) : css;
      out_css[c] = cssn;
      S += cssn;
    }
    shS = S;
  }
  __syncthreads();

  const float S = shS;
  const float denom = S + (float)C_DIM * 1e-5f;
  for (int e = tid; e < C_DIM * D_DIM; e += 256) {
    const int c = e >> 8;
    const float d  = decays[c];
    const bool  m  = (cs_accum[c] != 0.0f);
    const float cv = cv_in[e];
    const float cvn = m ? (cv * d + in_sum[e] * (1.0f - d)) : cv;
    out_cv[e] = cvn;
    const float cs = (out_css[c] + 1e-5f) / denom * S;
    out_dict[e] = m ? (cvn / cs) : cvn;
  }
}

// ---------- launch ----------
extern "C" void kernel_launch(void* const* d_in, const int* in_sizes, int n_in,
                              void* d_out, int out_size, void* d_ws, size_t ws_size,
                              hipStream_t stream) {
  const float* X      = (const float*)d_in[0];   // [N, 256]
  const float* dictf  = (const float*)d_in[1];   // [512, 256]
  const float* css_in = (const float*)d_in[2];   // [512]
  const float* cv_in  = (const float*)d_in[3];   // [512, 256]
  const float* cun_in = (const float*)d_in[4];   // [512]

  const int N = in_sizes[0] / D_DIM;

  // workspace layout
  char* ws = (char*)d_ws;
  unsigned short* dictb = (unsigned short*)ws;              // 512*256*2 = 262144
  float* d2       = (float*)(ws + 262144);                  // 2048
  float* cs_accum = (float*)(ws + 264192);                  // 2048
  float* in_sum   = (float*)(ws + 266240);                  // 524288
  float* decays   = (float*)(ws + 790528);                  // 2048

  // output layout (flat, reference return order)
  float* out_ge   = (float*)d_out;                          // N*256
  float* out_tk   = out_ge + (size_t)N * D_DIM;             // N*4
  float* out_dict = out_tk + (size_t)N * KTOP;              // 512*256
  float* out_css  = out_dict + (size_t)C_DIM * D_DIM;       // 512
  float* out_cv   = out_css + C_DIM;                        // 512*256
  float* out_cun  = out_cv + (size_t)C_DIM * D_DIM;         // 512

  hipMemsetAsync(cs_accum, 0, C_DIM * sizeof(float), stream);
  hipMemsetAsync(in_sum,   0, (size_t)C_DIM * D_DIM * sizeof(float), stream);

  prep_dict_kernel<<<C_DIM, 256, 0, stream>>>(dictf, dictb, d2);
  assign_kernel<<<N / MROWS, 256, 0, stream>>>(X, dictf, dictb, d2,
                                               out_ge, out_tk, cs_accum, in_sum);
  finalize_kernel<<<1, 256, 0, stream>>>(css_in, cv_in, cun_in, cs_accum, in_sum,
                                         out_dict, out_css, out_cv, out_cun, decays);
}
